// GatedLSTM_64398739636543
// MI455X (gfx1250) — matrix operand order repaired and checked by reading, not compiled, not run on hardware
//
#include <hip/hip_runtime.h>
#include <hip/hip_bf16.h>

// GatedLSTM  B=64 S=512 I=H=1024
// Phase 1: Xg = x @ W_gᵀ + b_g  (g = f,o,c) via bf16 WMMA, fp32 accum -> ws
// Phase 2: persistent cooperative scan, one grid barrier per step,
//          h ping-pong in bf16, c state in registers, all 3 gate GEMMs per wave.

#define HDIM   1024
#define IDIM   1024
#define BATCH  64
#define SEQ    512
#define MROWS  (BATCH * SEQ)                     // 32768
#define GATE_STRIDE ((size_t)MROWS * HDIM)       // 33,554,432 elements

typedef __attribute__((ext_vector_type(16))) __bf16        bf16x16;
typedef __attribute__((ext_vector_type(8)))  float         f32x8;
typedef __attribute__((ext_vector_type(4)))  unsigned int  u32x4;

union Frag { bf16x16 v; u32x4 q[2]; };

__device__ __forceinline__ f32x8 zero8() {
  f32x8 z = {0.f, 0.f, 0.f, 0.f, 0.f, 0.f, 0.f, 0.f};
  return z;
}

__device__ __forceinline__ f32x8 wmma_bf16(bf16x16 a, bf16x16 b, f32x8 c) {
  // (neg_a, A, neg_b, B, c_mod, C, reuse_a, reuse_b)
  return __builtin_amdgcn_wmma_f32_16x16x32_bf16(false, a, false, b, (short)0, c,
                                                 false, false);
}

// A (M=16,K=32 bf16) fragment: row-major source, `row` already points at the
// lane's row. Lanes 0-15: k = {0..7,16..23}; lanes 16-31: k = {8..15,24..31}.
__device__ __forceinline__ bf16x16 load_a(const unsigned short* __restrict__ row,
                                          int k, int hi8) {
  Frag f;
  f.q[0] = *(const u32x4*)(row + k + hi8);        // 8 bf16, 16B
  f.q[1] = *(const u32x4*)(row + k + 16 + hi8);   // 8 bf16, 16B
  return f.v;
}

// B (K=32,N=16 bf16) fragment: B[k][n] = W[n][k] (row-major W), lane's
// column n is W row n. Lanes 0-15: k = {0..7,16..23}; lanes 16-31: k = {8..15,24..31}.
__device__ __forceinline__ bf16x16 load_b(const unsigned short* __restrict__ wrow,
                                          int k, int khalf) {
  const int hi8 = khalf >> 1;
  Frag f;
  f.q[0] = *(const u32x4*)(wrow + k + hi8);
  f.q[1] = *(const u32x4*)(wrow + k + 16 + hi8);
  return f.v;
}

__device__ __forceinline__ unsigned short f2bf(float x) {
  unsigned u = __float_as_uint(x);
  u += 0x7FFFu + ((u >> 16) & 1u);   // round-to-nearest-even
  return (unsigned short)(u >> 16);
}

__device__ __forceinline__ float sigmoidf_(float x) {
  return 1.0f / (1.0f + __expf(-x));
}

// ---------------------------------------------------------------- conversions
__global__ __launch_bounds__(256) void cvt_bf16(const float* __restrict__ in,
                                                unsigned short* __restrict__ out,
                                                int n4) {
  int i = blockIdx.x * 256 + threadIdx.x;
  if (i < n4) {
    float4 v = ((const float4*)in)[i];
    unsigned lo = (unsigned)f2bf(v.x) | ((unsigned)f2bf(v.y) << 16);
    unsigned hi = (unsigned)f2bf(v.z) | ((unsigned)f2bf(v.w) << 16);
    ((uint2*)out)[i] = make_uint2(lo, hi);
  }
}

// ------------------------------------------------------- phase 1: Xg = xW^T+b
// Wave job: 4 M-tiles x 2 N-tiles (one gate), K-loop over IDIM.
// waves = 512 Mblocks * 96 Npairs = 49152; 4 waves/WG -> 12288 blocks of 128.
__global__ __launch_bounds__(128) void gemm_xw(
    const unsigned short* __restrict__ x16,   // [MROWS][IDIM] bf16
    const unsigned short* __restrict__ w16,   // [3][HDIM][IDIM] bf16 (f,o,c)
    const float* __restrict__ b_f, const float* __restrict__ b_o,
    const float* __restrict__ b_c,
    float* __restrict__ Xg)                   // [3][MROWS][HDIM] f32
{
  const int wave  = (blockIdx.x * blockDim.x + threadIdx.x) >> 5;
  const int lane  = threadIdx.x & 31;
  const int mblk  = wave / 96;          // 0..511  (4 M-tiles each)
  const int npair = wave % 96;          // 0..95   (2 N'-tiles each, N'=gate*64+nt)
  const int gate  = (2 * npair) / 64;   // 0,1,2 (never straddles: 64 is even)
  const int nt0   = (2 * npair) % 64;
  const int row16 = lane & 15;
  const int hi    = lane >> 4;
  const int hi8   = hi * 8;
  const int khalf = hi * 16;

  const unsigned short* arow[4];
#pragma unroll
  for (int i = 0; i < 4; ++i)
    arow[i] = x16 + (size_t)((mblk * 4 + i) * 16 + row16) * IDIM;
  const unsigned short* brow0 =
      w16 + ((size_t)gate * HDIM + nt0 * 16 + row16) * IDIM;
  const unsigned short* brow1 = brow0 + (size_t)16 * IDIM;

  f32x8 acc[4][2];
#pragma unroll
  for (int i = 0; i < 4; ++i) {
    acc[i][0] = zero8();
    acc[i][1] = zero8();
  }

  for (int k = 0; k < IDIM; k += 32) {
    bf16x16 b0 = load_b(brow0, k, khalf);
    bf16x16 b1 = load_b(brow1, k, khalf);
#pragma unroll
    for (int i = 0; i < 4; ++i) {
      bf16x16 a = load_a(arow[i], k, hi8);
      acc[i][0] = wmma_bf16(a, b0, acc[i][0]);
      acc[i][1] = wmma_bf16(a, b1, acc[i][1]);
    }
  }

  const float* bias = (gate == 0) ? b_f : (gate == 1) ? b_o : b_c;
#pragma unroll
  for (int j = 0; j < 2; ++j) {
    int n = (nt0 + j) * 16 + row16;
    float bv = bias[n];
#pragma unroll
    for (int i = 0; i < 4; ++i) {
#pragma unroll
      for (int r = 0; r < 8; ++r) {
        int m = (mblk * 4 + i) * 16 + hi8 + r;
        Xg[(size_t)gate * GATE_STRIDE + (size_t)m * HDIM + n] = acc[i][j][r] + bv;
      }
    }
  }
}

// ---------------------------------------------------------------- grid barrier
__device__ __forceinline__ void grid_barrier(unsigned* cnt, unsigned* gen,
                                             unsigned nwg) {
  __threadfence();   // release my stores to agent scope
  __syncthreads();
  if (threadIdx.x == 0) {
    unsigned g = __hip_atomic_load(gen, __ATOMIC_RELAXED,
                                   __HIP_MEMORY_SCOPE_AGENT);
    unsigned arrived = __hip_atomic_fetch_add(cnt, 1u, __ATOMIC_ACQ_REL,
                                              __HIP_MEMORY_SCOPE_AGENT);
    if (arrived == nwg - 1u) {
      __hip_atomic_store(cnt, 0u, __ATOMIC_RELAXED, __HIP_MEMORY_SCOPE_AGENT);
      __hip_atomic_store(gen, g + 1u, __ATOMIC_RELEASE,
                         __HIP_MEMORY_SCOPE_AGENT);
    } else {
      while (__hip_atomic_load(gen, __ATOMIC_ACQUIRE,
                               __HIP_MEMORY_SCOPE_AGENT) == g) {
        __builtin_amdgcn_s_sleep(2);
      }
    }
  }
  __syncthreads();
  __threadfence();   // acquire: make other WGs' stores visible
}

// ----------------------------------------------------- phase 2: recurrent scan
// 128 blocks x 64 threads = 256 waves; wave -> one 16(batch)x16(H) patch,
// all 3 gates. c state lives in registers across all 512 steps; h ping-pongs
// through bf16 global buffers, one grid barrier per step.
__global__ __launch_bounds__(64) void lstm_scan(
    const unsigned short* __restrict__ u16,   // [3][HDIM][HDIM] bf16 (f,o,c)
    const float* __restrict__ Xg,             // [3][MROWS][HDIM] f32
    unsigned short* __restrict__ hbuf,        // [2][BATCH][HDIM] bf16 (zeroed)
    float* __restrict__ out,                  // [BATCH*SEQ*HDIM] f32
    float* __restrict__ hc_final,             // h[B*H] then c[B*H]
    unsigned* __restrict__ bar)               // bar[0]=count, bar[32]=gen
{
  const int wave  = (blockIdx.x << 1) | (threadIdx.x >> 5);  // 0..255
  const int lane  = threadIdx.x & 31;
  const int mtile = wave >> 6;    // 0..3   (batch tiles)
  const int ntile = wave & 63;    // 0..63  (H tiles)
  const int row16 = lane & 15;
  const int hi    = lane >> 4;
  const int hi8   = hi * 8;
  const int khalf = hi * 16;
  const int n     = ntile * 16 + row16;

  const unsigned short* ub[3];
#pragma unroll
  for (int g = 0; g < 3; ++g)
    ub[g] = u16 + ((size_t)g * HDIM + n) * HDIM;

  f32x8 cstate = zero8();
  unsigned* cnt = bar;
  unsigned* gen = bar + 32;  // separate cacheline

  for (int t = 0; t < SEQ; ++t) {
    const unsigned short* hin  = hbuf + (size_t)(t & 1) * (BATCH * HDIM);
    unsigned short*       hout = hbuf + (size_t)((t + 1) & 1) * (BATCH * HDIM);
    const unsigned short* arow = hin + (size_t)(mtile * 16 + row16) * HDIM;

    f32x8 af = zero8(), ao = zero8(), ac = zero8();
    for (int k = 0; k < HDIM; k += 32) {
      bf16x16 a = load_a(arow, k, hi8);
      af = wmma_bf16(a, load_b(ub[0], k, khalf), af);
      ao = wmma_bf16(a, load_b(ub[1], k, khalf), ao);
      ac = wmma_bf16(a, load_b(ub[2], k, khalf), ac);
    }

#pragma unroll
    for (int r = 0; r < 8; ++r) {
      int b = mtile * 16 + hi8 + r;
      size_t xrow = (size_t)(b * SEQ + t) * HDIM + n;
      float xf = Xg[xrow];
      float xo = Xg[GATE_STRIDE + xrow];
      float xc = Xg[2 * GATE_STRIDE + xrow];
      float fg = sigmoidf_(xf + af[r]);
      float og = sigmoidf_(xo + ao[r]);
      float cc = fg * cstate[r] + tanhf(xc + ac[r]);
      cstate[r] = cc;
      float h = og * cc;
      out[xrow] = h;                          // output[b][t][n]
      hout[(size_t)b * HDIM + n] = f2bf(h);   // h for next step's GEMM
      if (t == SEQ - 1) {
        hc_final[(size_t)b * HDIM + n] = h;
        hc_final[(size_t)BATCH * HDIM + (size_t)b * HDIM + n] = cc;
      }
    }
    grid_barrier(cnt, gen, gridDim.x);
  }
}

// ------------------------------------------------------------------- launcher
extern "C" void kernel_launch(void* const* d_in, const int* in_sizes, int n_in,
                              void* d_out, int out_size, void* d_ws,
                              size_t ws_size, hipStream_t stream) {
  const float* x   = (const float*)d_in[0];
  const float* W_f = (const float*)d_in[1];
  const float* U_f = (const float*)d_in[2];
  const float* b_f = (const float*)d_in[3];
  // d_in[4..6] = W_i/U_i/b_i: dead in the reference math
  const float* W_o = (const float*)d_in[7];
  const float* U_o = (const float*)d_in[8];
  const float* b_o = (const float*)d_in[9];
  const float* W_c = (const float*)d_in[10];
  const float* U_c = (const float*)d_in[11];
  const float* b_c = (const float*)d_in[12];

  char* ws = (char*)d_ws;
  const size_t OFF_X16 = 0;                                        // 67,108,864 B
  const size_t OFF_W16 = OFF_X16 + (size_t)MROWS * IDIM * 2;       // +6,291,456 B
  const size_t OFF_U16 = OFF_W16 + 3ull * HDIM * IDIM * 2;         // +6,291,456 B
  const size_t OFF_XG  = OFF_U16 + 3ull * HDIM * HDIM * 2;         // +402,653,184 B
  const size_t OFF_H   = OFF_XG + 3ull * GATE_STRIDE * 4;          // +262,144 B
  const size_t OFF_BAR = OFF_H + 2ull * BATCH * HDIM * 2;          // +256 B
  // total ~482.6 MB of workspace

  unsigned short* x16  = (unsigned short*)(ws + OFF_X16);
  unsigned short* w16  = (unsigned short*)(ws + OFF_W16);
  unsigned short* u16  = (unsigned short*)(ws + OFF_U16);
  float*          Xg   = (float*)(ws + OFF_XG);
  unsigned short* hbuf = (unsigned short*)(ws + OFF_H);
  unsigned*       bar  = (unsigned*)(ws + OFF_BAR);

  // Zero h ping-pong buffers (h0 = 0) and barrier words every call so graph
  // replays are deterministic (ws is poisoned once, never re-poisoned).
  hipMemsetAsync(ws + OFF_H, 0, 2ull * BATCH * HDIM * 2 + 256, stream);

  // fp32 -> bf16 conversions
  {
    int n4 = (int)((size_t)MROWS * IDIM / 4);
    cvt_bf16<<<(n4 + 255) / 256, 256, 0, stream>>>(x, x16, n4);
    int w4 = (int)((size_t)HDIM * IDIM / 4);
    int wgrid = (w4 + 255) / 256;
    cvt_bf16<<<wgrid, 256, 0, stream>>>(W_f, w16 + 0ull * HDIM * IDIM, w4);
    cvt_bf16<<<wgrid, 256, 0, stream>>>(W_o, w16 + 1ull * HDIM * IDIM, w4);
    cvt_bf16<<<wgrid, 256, 0, stream>>>(W_c, w16 + 2ull * HDIM * IDIM, w4);
    cvt_bf16<<<wgrid, 256, 0, stream>>>(U_f, u16 + 0ull * HDIM * HDIM, w4);
    cvt_bf16<<<wgrid, 256, 0, stream>>>(U_o, u16 + 1ull * HDIM * HDIM, w4);
    cvt_bf16<<<wgrid, 256, 0, stream>>>(U_c, u16 + 2ull * HDIM * HDIM, w4);
  }

  // Phase 1: 49152 waves (512 Mblocks x 96 Npairs), 4 waves per 128-thread WG.
  gemm_xw<<<12288, 128, 0, stream>>>(x16, w16, b_f, b_o, b_c, Xg);

  // Phase 2: persistent cooperative scan. 128 WGs x 64 threads (256 waves).
  float* out = (float*)d_out;
  lstm_scan<<<128, 64, 0, stream>>>(u16, Xg, hbuf, out,
                                    out + (size_t)MROWS * HDIM, bar);
}
